// DualQuerySelection_78546361909926
// MI455X (gfx1250) — matrix operand order repaired and checked by reading, not compiled, not run on hardware
//
#include <hip/hip_runtime.h>
#include <hip/hip_bf16.h>
#include <math.h>

typedef __attribute__((ext_vector_type(16))) _Float16 v16h;
typedef __attribute__((ext_vector_type(8)))  float    v8f;

#define BATCH 8
#define CHN   256
#define HGT   200
#define WID   200
#define HW    40000          // HGT*WID
#define DHID  128
#define NSEL  1000

// d_out layout (floats): sel[8*1000*256] | query_pos[8*1000*3] | fg_logits[8*40000] | quality[8*1000]
#define SEL_OFF 0
#define QP_OFF  (8*1000*256)
#define FG_OFF  (QP_OFF + 8*1000*3)
#define QU_OFF  (FG_OFF + 8*HW)

// ---------- helpers ----------

// bin 0 == largest float (order-preserving uint key, top 10 bits)
__device__ __forceinline__ unsigned binOf(float f) {
    unsigned u = __float_as_uint(f);
    unsigned key = (u & 0x80000000u) ? ~u : (u | 0x80000000u);
    return 1023u - (key >> 22);
}

// Pre-pack a (256 x 128) fp32 weight matrix into WMMA A-fragments (f16) in LDS.
// sA[kk][t][lane] = v16h fragment for K-block kk, hidden-tile t.
// A k order used here (contiguous, same as B): lanes 0-15: M=lane, K = 0..15; lanes 16-31: M=lane-16, K = 16..31.
__device__ __forceinline__ void fill_wfrag(v16h (*sA)[8][32], const float* w) {
    _Float16* sp = (_Float16*)sA;
    for (int e = threadIdx.x; e < 8 * 8 * 32 * 16; e += blockDim.x) {
        int j    = e & 15;
        int lane = (e >> 4) & 31;
        int t    = (e >> 9) & 7;
        int kk   = e >> 12;
        int m     = lane & 15;
        int base0 = (lane < 16) ? 0 : 16;
        int kl    = base0 + j;
        sp[e] = (_Float16)w[(kk * 32 + kl) * DHID + t * 16 + m];
    }
}

// ---------- stage 1: fused (bev+pos) -> MLP -> fg_logits, WMMA f16 ----------
__global__ __launch_bounds__(128) void fg_mlp_kernel(
    const float* __restrict__ bev, const float* __restrict__ pos,
    const float* __restrict__ w1,  const float* __restrict__ b1,
    const float* __restrict__ w2,  const float* __restrict__ b2,
    float* __restrict__ fgOut)
{
    __shared__ v16h sA[8][8][32];           // 64 KB: all of w1, fragment-packed
    fill_wfrag(sA, w1);
    __syncthreads();

    int wave = threadIdx.x >> 5;
    int lane = threadIdx.x & 31;
    int b    = blockIdx.y;
    int row0 = (blockIdx.x * 4 + wave) * 16;        // 2500 tiles/batch, exact
    int r    = row0 + (lane & 15);

    const float* bp = bev + (size_t)b * CHN * HW;
    const float* pp = pos + (size_t)b * CHN * HW;

    v8f acc[8] = {};
    for (int kk = 0; kk < 8; kk++) {
        // B fragment: xT tile, 32ch x 16rows.  lanes 0-15: K=kb..kb+15 (N=lane); lanes 16-31: K=kb+16..kb+31
        int kb = kk * 32 + ((lane < 16) ? 0 : 16);
        v16h bf;
#pragma unroll
        for (int j = 0; j < 8; j++) {
            int k0 = kb + 2 * j;
            float v0 = bp[(size_t)k0 * HW + r] + pp[(size_t)k0 * HW + r];
            float v1 = bp[(size_t)(k0 + 1) * HW + r] + pp[(size_t)(k0 + 1) * HW + r];
            bf[2 * j]     = (_Float16)v0;
            bf[2 * j + 1] = (_Float16)v1;
        }
#pragma unroll
        for (int t = 0; t < 8; t++) {
            v16h af = sA[kk][t][lane];
            acc[t] = __builtin_amdgcn_wmma_f32_16x16x32_f16(
                false, af, false, bf, (short)0, acc[t], false, false);
        }
    }

    // bias + ReLU + second-layer dot (w2 is 128x1).  D layout: lane n holds M=msel..msel+7.
    int msel = (lane < 16) ? 0 : 8;
    float partial = 0.f;
#pragma unroll
    for (int t = 0; t < 8; t++) {
#pragma unroll
        for (int m = 0; m < 8; m++) {
            int h = t * 16 + msel + m;
            float v = fmaxf(acc[t][m] + b1[h], 0.f);
            partial = fmaf(v, w2[h], partial);
        }
    }
    partial += __shfl_xor(partial, 16);
    if (lane < 16)
        fgOut[(size_t)b * HW + r] = partial + b2[0];
}

// ---------- stage 2: deterministic per-batch top-1000 ----------
__global__ void zero_hist_kernel(unsigned* hist) {
    int i = blockIdx.x * blockDim.x + threadIdx.x;
    if (i < BATCH * 1024) hist[i] = 0u;
}

__global__ void hist_kernel(const float* __restrict__ fg, unsigned* __restrict__ hist) {
    int b = blockIdx.y;
    int i = blockIdx.x * blockDim.x + threadIdx.x;
    if (i < HW) atomicAdd(&hist[b * 1024 + binOf(fg[(size_t)b * HW + i])], 1u);
}

__global__ void thresh_kernel(const unsigned* __restrict__ hist, int* __restrict__ thr) {
    int b = blockIdx.x;
    if (threadIdx.x == 0) {
        int cum = 0, T = 1023, before = 0;
        for (int i = 0; i < 1024; i++) {
            int c = (int)hist[b * 1024 + i];
            if (cum + c >= NSEL) { T = i; before = cum; break; }
            cum += c;
        }
        thr[b * 4 + 0] = T;
        thr[b * 4 + 1] = NSEL - before;   // rem taken from threshold bin
        thr[b * 4 + 2] = before;          // count strictly above threshold
    }
}

__global__ __launch_bounds__(1024) void compact_kernel(
    const float* __restrict__ fg, const int* __restrict__ thr, int* __restrict__ topk)
{
    int b = blockIdx.x;
    unsigned T = (unsigned)thr[b * 4 + 0];
    int rem      = thr[b * 4 + 1];
    int cntAbove = thr[b * 4 + 2];

    __shared__ unsigned s_hi[33], s_eq[33];
    __shared__ int runHi, runEq;
    int tid = threadIdx.x, lane = tid & 31, wv = tid >> 5;
    if (tid == 0) { runHi = 0; runEq = 0; }
    __syncthreads();

    const float* f = fg + (size_t)b * HW;
    for (int base = 0; base < HW; base += 1024) {
        int idx = base + tid;
        bool valid = idx < HW;
        unsigned bin = valid ? binOf(f[idx]) : 0xFFFFFFFFu;
        bool hi = valid && (bin < T);
        bool eq = valid && (bin == T);
        unsigned mhi = __builtin_amdgcn_ballot_w32(hi);
        unsigned meq = __builtin_amdgcn_ballot_w32(eq);
        if (lane == 0) { s_hi[wv] = (unsigned)__builtin_popcount(mhi);
                         s_eq[wv] = (unsigned)__builtin_popcount(meq); }
        __syncthreads();
        if (tid == 0) {                       // exclusive scan across 32 waves
            unsigned ch = 0, ce = 0;
            for (int i = 0; i < 32; i++) {
                unsigned th = s_hi[i], te = s_eq[i];
                s_hi[i] = ch; s_eq[i] = ce; ch += th; ce += te;
            }
            s_hi[32] = ch; s_eq[32] = ce;
        }
        __syncthreads();
        unsigned lmask = (1u << lane) - 1u;
        int offH = runHi + (int)s_hi[wv] + __builtin_popcount(mhi & lmask);
        int offE = runEq + (int)s_eq[wv] + __builtin_popcount(meq & lmask);
        if (hi)                 topk[b * 1024 + offH] = idx;
        if (eq && offE < rem)   topk[b * 1024 + cntAbove + offE] = idx;
        __syncthreads();
        if (tid == 0) { runHi += (int)s_hi[32]; runEq += (int)s_eq[32]; }
        __syncthreads();
    }
}

// ---------- stage 3: gather sel rows, quality & pos heads (WMMA x2) ----------
__global__ __launch_bounds__(128) void sel_mlp_kernel(
    const float* __restrict__ bev, const float* __restrict__ pos,
    const float* __restrict__ qw1, const float* __restrict__ qb1,
    const float* __restrict__ qw2, const float* __restrict__ qb2,
    const float* __restrict__ pw1, const float* __restrict__ pb1,
    const float* __restrict__ pw2, const float* __restrict__ pb2,
    const int* __restrict__ topk,
    float* __restrict__ selOut, float* __restrict__ qpOut, float* __restrict__ quOut)
{
    __shared__ v16h sA[8][8][32];           // 64 KB, refilled between passes
    int wave = threadIdx.x >> 5;
    int lane = threadIdx.x & 31;
    int b    = blockIdx.y;
    int tile = blockIdx.x * 4 + wave;       // 63 tiles/batch cover 1008 >= 1000
    bool active = tile < 63;
    int q  = tile * 16 + (lane & 15);
    bool qv = active && (q < NSEL);
    int idx = qv ? topk[b * 1024 + q] : 0;

    const float* bp = bev + (size_t)b * CHN * HW;
    const float* pp = pos + (size_t)b * CHN * HW;
    float* selRow = selOut + SEL_OFF + ((size_t)b * NSEL + q) * CHN;
    int msel = (lane < 16) ? 0 : 8;

    // ---- pass 1: quality head (also emits sel rows during the gather) ----
    fill_wfrag(sA, qw1);
    __syncthreads();
    float pq = 0.f;
    if (active) {
        v8f acc[8] = {};
        for (int kk = 0; kk < 8; kk++) {
            int kb = kk * 32 + ((lane < 16) ? 0 : 16);
            v16h bf;
#pragma unroll
            for (int j = 0; j < 8; j++) {
                int k0 = kb + 2 * j;
                float v0 = bp[(size_t)k0 * HW + idx] + pp[(size_t)k0 * HW + idx];
                float v1 = bp[(size_t)(k0 + 1) * HW + idx] + pp[(size_t)(k0 + 1) * HW + idx];
                if (qv) { selRow[k0] = v0; selRow[k0 + 1] = v1; }
                bf[2 * j]     = (_Float16)v0;
                bf[2 * j + 1] = (_Float16)v1;
            }
#pragma unroll
            for (int t = 0; t < 8; t++) {
                v16h af = sA[kk][t][lane];
                acc[t] = __builtin_amdgcn_wmma_f32_16x16x32_f16(
                    false, af, false, bf, (short)0, acc[t], false, false);
            }
        }
#pragma unroll
        for (int t = 0; t < 8; t++)
#pragma unroll
            for (int m = 0; m < 8; m++) {
                int h = t * 16 + msel + m;
                float v = fmaxf(acc[t][m] + qb1[h], 0.f);
                pq = fmaf(v, qw2[h], pq);
            }
        pq += __shfl_xor(pq, 16);
        if (qv && lane < 16)
            quOut[QU_OFF + b * NSEL + q] = 1.f / (1.f + expf(-(pq + qb2[0])));
    }
    __syncthreads();

    // ---- pass 2: position head ----
    fill_wfrag(sA, pw1);
    __syncthreads();
    if (active) {
        v8f acc[8] = {};
        for (int kk = 0; kk < 8; kk++) {
            int kb = kk * 32 + ((lane < 16) ? 0 : 16);
            v16h bf;
#pragma unroll
            for (int j = 0; j < 8; j++) {
                int k0 = kb + 2 * j;
                float v0 = bp[(size_t)k0 * HW + idx] + pp[(size_t)k0 * HW + idx];
                float v1 = bp[(size_t)(k0 + 1) * HW + idx] + pp[(size_t)(k0 + 1) * HW + idx];
                bf[2 * j]     = (_Float16)v0;
                bf[2 * j + 1] = (_Float16)v1;
            }
#pragma unroll
            for (int t = 0; t < 8; t++) {
                v16h af = sA[kk][t][lane];
                acc[t] = __builtin_amdgcn_wmma_f32_16x16x32_f16(
                    false, af, false, bf, (short)0, acc[t], false, false);
            }
        }
        float p0 = 0.f, p1 = 0.f, p2 = 0.f;
#pragma unroll
        for (int t = 0; t < 8; t++)
#pragma unroll
            for (int m = 0; m < 8; m++) {
                int h = t * 16 + msel + m;
                float v = fmaxf(acc[t][m] + pb1[h], 0.f);
                p0 = fmaf(v, pw2[h * 3 + 0], p0);
                p1 = fmaf(v, pw2[h * 3 + 1], p1);
                p2 = fmaf(v, pw2[h * 3 + 2], p2);
            }
        p0 += __shfl_xor(p0, 16);
        p1 += __shfl_xor(p1, 16);
        p2 += __shfl_xor(p2, 16);
        if (qv && lane < 16) {
            float xo = tanhf(p0 + pb2[0]) * 5.f;
            float yo = tanhf(p1 + pb2[1]) * 5.f;
            float zo = tanhf(p2 + pb2[2]) * 5.f;
            int xi = idx % WID, yi = idx / WID;
            float xw = -51.2f + ((xi + 0.5f) / (float)WID) * 102.4f;
            float yw = -51.2f + ((yi + 0.5f) / (float)HGT) * 102.4f;
            float* o = qpOut + QP_OFF + ((size_t)b * NSEL + q) * 3;
            o[0] = xw + xo; o[1] = yw + yo; o[2] = zo;
        }
    }
}

// ---------- launch ----------
extern "C" void kernel_launch(void* const* d_in, const int* in_sizes, int n_in,
                              void* d_out, int out_size, void* d_ws, size_t ws_size,
                              hipStream_t stream) {
    const float* bev = (const float*)d_in[0];
    const float* pos = (const float*)d_in[1];
    const float* fg_w1 = (const float*)d_in[2];
    const float* fg_b1 = (const float*)d_in[3];
    const float* fg_w2 = (const float*)d_in[4];
    const float* fg_b2 = (const float*)d_in[5];
    const float* q_w1  = (const float*)d_in[6];
    const float* q_b1  = (const float*)d_in[7];
    const float* q_w2  = (const float*)d_in[8];
    const float* q_b2  = (const float*)d_in[9];
    const float* p_w1  = (const float*)d_in[10];
    const float* p_b1  = (const float*)d_in[11];
    const float* p_w2  = (const float*)d_in[12];
    const float* p_b2  = (const float*)d_in[13];

    float* out = (float*)d_out;
    float* fgOut = out + FG_OFF;

    // workspace: hist[8][1024] u32 | thr[8][4] i32 | topk[8][1024] i32
    unsigned* hist = (unsigned*)d_ws;
    int* thr  = (int*)d_ws + BATCH * 1024;
    int* topk = thr + BATCH * 4;

    zero_hist_kernel<<<32, 256, 0, stream>>>(hist);
    fg_mlp_kernel<<<dim3(625, BATCH), 128, 0, stream>>>(
        bev, pos, fg_w1, fg_b1, fg_w2, fg_b2, fgOut);
    hist_kernel<<<dim3((HW + 255) / 256, BATCH), 256, 0, stream>>>(fgOut, hist);
    thresh_kernel<<<BATCH, 32, 0, stream>>>(hist, thr);
    compact_kernel<<<BATCH, 1024, 0, stream>>>(fgOut, thr, topk);
    sel_mlp_kernel<<<dim3(16, BATCH), 128, 0, stream>>>(
        bev, pos, q_w1, q_b1, q_w2, q_b2, p_w1, p_b1, p_w2, p_b2,
        topk, out, out, out);
}
